// AttnRawDecoderWithSrc_12756052869524
// MI455X (gfx1250) — matrix operand order repaired and checked by reading, not compiled, not run on hardware
//
#include <hip/hip_runtime.h>
#include <hip/hip_bf16.h>
#include <math.h>

// ---------------- dimensions ----------------
#define S        256
#define B        32
#define EMB      256
#define ENC_IN   256
#define ENC_E    512
#define LSTM     512
#define NL       3
#define G4       2048           // 4*LSTM
#define SHRINK   512
#define VOCAB    10000
#define VOCABP   10016          // padded to multiple of 32
#define ROWS     (S * B)        // 8192
#define LOGITS_N ((size_t)ROWS * VOCAB)

typedef __attribute__((ext_vector_type(16))) _Float16 v16h;
typedef __attribute__((ext_vector_type(8)))  _Float16 v8h;
typedef __attribute__((ext_vector_type(8)))  float    v8f;

// ---------------- WMMA fragment helpers (CDNA5 16x16x32 f16 layouts) ------
// A matrix 16x32 (MxK), row-major A with leading dim lda:
//   lane L: row m = m0 + (L&15), hi = L>>4
//   element j: k = k0 + j + 16*hi   -> one contiguous 32B load per lane
__device__ __forceinline__ v16h frag_a(const _Float16* __restrict__ A, int lda,
                                       int m0, int k0, int lane) {
    int m  = m0 + (lane & 15);
    int hi = lane >> 4;
    return *(const v16h*)(A + (size_t)m * lda + k0 + hi * 16);
}

// B matrix 32x16 (KxN) sourced from row-major W of shape (N,K), leading dim ldw:
//   lane L: col n = n0 + (L&15), hi = L>>4
//   element j: k = k0 + j + 16*hi -> one contiguous 32B load per lane
__device__ __forceinline__ v16h frag_b(const _Float16* __restrict__ W, int ldw,
                                       int n0, int k0, int lane) {
    int n  = n0 + (lane & 15);
    int hi = lane >> 4;
    return *(const v16h*)(W + (size_t)n * ldw + k0 + hi * 16);
}

__device__ __forceinline__ float sigf(float x) { return 1.0f / (1.0f + __expf(-x)); }

// ---------------- 2x2-blocked WMMA GEMM:  D = A(MxK) * W(NxK)^T + bias ----
// Each wave owns a 32x32 output block (2 M-tiles x 2 N-tiles): per k-step it
// loads 2 A-frags + 2 B-frags and issues 4 WMMAs -> ~16 FLOP/byte from L2.
// 8 waves per 256-thread workgroup. M,N,K multiples of 32.
__global__ void gemm_wmma(const _Float16* __restrict__ A,
                          const _Float16* __restrict__ W,
                          const float*    __restrict__ bias,
                          float*          __restrict__ Df32,
                          _Float16*       __restrict__ Df16,
                          int M, int N, int K, int ldd, int nmax) {
    const int wave = threadIdx.x >> 5;
    const int lane = threadIdx.x & 31;
    const int tn   = N >> 5;
    const int tile = blockIdx.x * 8 + wave;
    if (tile >= (M >> 5) * tn) return;
    const int m0 = (tile / tn) << 5;
    const int n0 = (tile % tn) << 5;

    v8f acc00 = {}, acc01 = {}, acc10 = {}, acc11 = {};
    for (int k0 = 0; k0 < K; k0 += 32) {
        v16h a0 = frag_a(A, K, m0,      k0, lane);
        v16h a1 = frag_a(A, K, m0 + 16, k0, lane);
        v16h b0 = frag_b(W, K, n0,      k0, lane);
        v16h b1 = frag_b(W, K, n0 + 16, k0, lane);
        acc00 = __builtin_amdgcn_wmma_f32_16x16x32_f16(false, a0, false, b0, (short)0, acc00, false, false);
        acc01 = __builtin_amdgcn_wmma_f32_16x16x32_f16(false, a0, false, b1, (short)0, acc01, false, false);
        acc10 = __builtin_amdgcn_wmma_f32_16x16x32_f16(false, a1, false, b0, (short)0, acc10, false, false);
        acc11 = __builtin_amdgcn_wmma_f32_16x16x32_f16(false, a1, false, b1, (short)0, acc11, false, false);
    }
    const int hi = lane >> 4;
    const int nn = lane & 15;
    #pragma unroll
    for (int bn = 0; bn < 2; ++bn) {
        const int n = n0 + bn * 16 + nn;
        if (n >= nmax) continue;
        const float bv = bias ? bias[n] : 0.0f;
        const v8f* accs[2] = { bn ? &acc01 : &acc00, bn ? &acc11 : &acc10 };
        #pragma unroll
        for (int bm = 0; bm < 2; ++bm) {
            const v8f& a = *accs[bm];
            const int  mb = m0 + bm * 16 + hi * 8;
            if (Df16) {
                for (int r = 0; r < 8; ++r)
                    Df16[(size_t)(mb + r) * ldd + n] = (_Float16)(a[r] + bv);
            } else {
                for (int r = 0; r < 8; ++r)
                    Df32[(size_t)(mb + r) * ldd + n] = a[r] + bv;
            }
        }
    }
}

// ---------------- LSTM recurrent step -------------------------------------
// gates = Gt(32x2048, precomputed x*Wih^T + bias) + h_prev * Whh^T
// 16 workgroups; WG wg owns 32 gate-columns j0=wg*32 across all 4 gates.
// 8 waves: wave w -> gate g=w>>1, m-tile mt=w&1; each wave does 2 N-tiles
// (n0, n0+16) reusing its A fragment (h_prev rows).
__global__ void lstm_step(const _Float16* __restrict__ Hprev,
                          const _Float16* __restrict__ Whh,
                          const float*    __restrict__ Gt,
                          float*          __restrict__ Cst,
                          _Float16*       __restrict__ Hnext,
                          _Float16*       __restrict__ Xnext,
                          float* __restrict__ outH, float* __restrict__ outC,
                          int isLast) {
    __shared__ float gl[4][B][32];
    const int lane = threadIdx.x & 31;
    const int wave = threadIdx.x >> 5;
    const int gate = wave >> 1;
    const int m0   = (wave & 1) << 4;
    const int j0   = blockIdx.x << 5;
    const int n0   = gate * LSTM + j0;

    v8f acc0 = {}, acc1 = {};
    for (int k0 = 0; k0 < LSTM; k0 += 32) {
        v16h a  = frag_a(Hprev, LSTM, m0, k0, lane);
        v16h b0 = frag_b(Whh, LSTM, n0,      k0, lane);
        v16h b1 = frag_b(Whh, LSTM, n0 + 16, k0, lane);
        acc0 = __builtin_amdgcn_wmma_f32_16x16x32_f16(false, a, false, b0, (short)0, acc0, false, false);
        acc1 = __builtin_amdgcn_wmma_f32_16x16x32_f16(false, a, false, b1, (short)0, acc1, false, false);
    }
    const int hi = lane >> 4;
    const int nn = lane & 15;
    #pragma unroll
    for (int bn = 0; bn < 2; ++bn) {
        const v8f& a = bn ? acc1 : acc0;
        const int  n = n0 + bn * 16 + nn;
        const int  jl = bn * 16 + nn;
        for (int r = 0; r < 8; ++r) {
            int m = m0 + r + hi * 8;
            gl[gate][m][jl] = a[r] + Gt[(size_t)m * G4 + n];
        }
    }
    __syncthreads();

    // 1024 elements (32 rows x 32 cols) per WG -> 4 per thread
    for (int e = threadIdx.x; e < B * 32; e += blockDim.x) {
        const int b  = e >> 5;
        const int jj = e & 31;
        const int jg = j0 + jj;
        const float gi = gl[0][b][jj];
        const float gf = gl[1][b][jj];
        const float gg = gl[2][b][jj];
        const float go = gl[3][b][jj];
        const size_t si = (size_t)b * LSTM + jg;
        const float cp = Cst[si];
        const float cn = sigf(gf) * cp + sigf(gi) * tanhf(gg);
        const float hn = sigf(go) * tanhf(cn);
        Cst[si]   = cn;
        Hnext[si] = (_Float16)hn;
        Xnext[si] = (_Float16)hn;
        if (isLast) { outH[si] = hn; outC[si] = cn; }
    }
}

// ---------------- windowed attention (one wave per (t,b)) -----------------
__global__ void attn_kernel(const float* __restrict__ enc_outputs,
                            const float* __restrict__ qbuf,
                            const _Float16* __restrict__ dec,
                            _Float16* __restrict__ att) {
    const int wid  = blockIdx.x * 8 + (threadIdx.x >> 5);
    const int lane = threadIdx.x & 31;
    if (wid >= ROWS) return;
    const int t = wid >> 5;
    const int b = wid & 31;

    const float* qp = qbuf + (size_t)wid * ENC_E;
    const float* wp[6];
    float sc[6];
    for (int w = 0; w < 6; ++w) {
        int src = t - 3 + w;
        int scl = src < 0 ? 0 : (src > S - 1 ? S - 1 : src);
        wp[w] = enc_outputs + ((size_t)scl * B + b) * ENC_E;
        float p = 0.0f;
        for (int e = lane; e < ENC_E; e += 32) p += wp[w][e] * qp[e];
        for (int off = 16; off; off >>= 1) p += __shfl_xor(p, off, 32);
        sc[w] = (src >= 0 && src < S) ? p : -1e30f;
    }
    float mx = sc[0];
    for (int w = 1; w < 6; ++w) mx = fmaxf(mx, sc[w]);
    float den = 0.0f, al[6];
    for (int w = 0; w < 6; ++w) { al[w] = __expf(sc[w] - mx); den += al[w]; }
    const float inv = 1.0f / den;
    for (int w = 0; w < 6; ++w) al[w] *= inv;

    _Float16* arow = att + (size_t)wid * (LSTM + ENC_E);
    const _Float16* drow = dec + (size_t)wid * LSTM;
    for (int e = lane; e < ENC_E; e += 32) {
        float a = 0.0f;
        for (int w = 0; w < 6; ++w) a += al[w] * wp[w][e];
        arow[LSTM + e] = (_Float16)a;
        arow[e]        = drow[e];       // concat [dec, ctx]
    }
}

// ---------------- small utility kernels -----------------------------------
__global__ void cvt_f16(const float* __restrict__ s, _Float16* __restrict__ d, int n) {
    int i = blockIdx.x * blockDim.x + threadIdx.x;
    if (i < n) d[i] = (_Float16)s[i];
}
__global__ void cvt_outw(const float* __restrict__ s, _Float16* __restrict__ d) {
    int i = blockIdx.x * blockDim.x + threadIdx.x;      // over VOCABP*512
    if (i >= VOCABP * SHRINK) return;
    int n = i >> 9, k = i & 511;
    d[i] = (n < VOCAB) ? (_Float16)s[(size_t)n * SHRINK + k] : (_Float16)0.0f;
}
__global__ void combine_bias(const float* __restrict__ bih, const float* __restrict__ bhh,
                             float* __restrict__ d, int n) {
    int i = blockIdx.x * blockDim.x + threadIdx.x;
    if (i < n) d[i] = bih[i] + bhh[i];
}
__global__ void build_x0(const int* __restrict__ idx, const float* __restrict__ emb,
                         const float* __restrict__ encin, _Float16* __restrict__ X0) {
    int i = blockIdx.x * blockDim.x + threadIdx.x;      // over ROWS*512
    if (i >= ROWS * 512) return;
    int row = i >> 9, k = i & 511;
    float v = (k < EMB) ? emb[(size_t)idx[row] * EMB + k]
                        : encin[(size_t)row * ENC_IN + (k - EMB)];
    X0[i] = (_Float16)v;
}
__global__ void init_state(const float* __restrict__ h0, const float* __restrict__ c0,
                           _Float16* __restrict__ H, float* __restrict__ C) {
    int i = blockIdx.x * blockDim.x + threadIdx.x;      // over B*LSTM
    if (i >= B * LSTM) return;
    H[i] = (_Float16)h0[i];
    C[i] = c0[i];
}

// ---------------- host orchestration --------------------------------------
static inline size_t alignup(size_t x) { return (x + 255) & ~(size_t)255; }

extern "C" void kernel_launch(void* const* d_in, const int* in_sizes, int n_in,
                              void* d_out, int out_size, void* d_ws, size_t ws_size,
                              hipStream_t stream) {
    (void)in_sizes; (void)n_in; (void)out_size; (void)ws_size;
    const int*   idx    = (const int*)  d_in[0];
    const float* h0     = (const float*)d_in[1];
    const float* c0     = (const float*)d_in[2];
    const float* enco   = (const float*)d_in[3];
    const float* enci   = (const float*)d_in[4];
    const float* embt   = (const float*)d_in[5];
    const float* wih    = (const float*)d_in[6];
    const float* whh    = (const float*)d_in[7];
    const float* bih    = (const float*)d_in[8];
    const float* bhh    = (const float*)d_in[9];
    const float* attW   = (const float*)d_in[10];
    const float* attb   = (const float*)d_in[11];
    const float* shrW   = (const float*)d_in[12];
    const float* shrb   = (const float*)d_in[13];
    const float* outW   = (const float*)d_in[14];
    const float* outb   = (const float*)d_in[15];
    float* out = (float*)d_out;

    // ---- workspace carve ----
    char* p = (char*)d_ws;  size_t off = 0;
    auto carve = [&](size_t bytes) { char* r = p + off; off = alignup(off + bytes); return r; };
    _Float16* wih16 = (_Float16*)carve((size_t)NL * G4 * LSTM * 2);
    _Float16* whh16 = (_Float16*)carve((size_t)NL * G4 * LSTM * 2);
    _Float16* atw16 = (_Float16*)carve((size_t)ENC_E * LSTM * 2);
    _Float16* shw16 = (_Float16*)carve((size_t)SHRINK * (LSTM + ENC_E) * 2);
    _Float16* otw16 = (_Float16*)carve((size_t)VOCABP * SHRINK * 2);
    float*    biasc = (float*)   carve((size_t)NL * G4 * 4);
    _Float16* Xa    = (_Float16*)carve((size_t)ROWS * LSTM * 2);
    _Float16* Xb    = (_Float16*)carve((size_t)ROWS * LSTM * 2);
    _Float16* Hp0   = (_Float16*)carve((size_t)B * LSTM * 2);
    _Float16* Hp1   = (_Float16*)carve((size_t)B * LSTM * 2);
    float*    Cst   = (float*)   carve((size_t)B * LSTM * 4);
    char*     Gbase = carve((size_t)ROWS * G4 * 4);      // 64 MB, reused below
    float*    Gbuf  = (float*)Gbase;
    // after LSTM finishes, reuse G region for attention scratch:
    float*    qbuf  = (float*)Gbase;                                  // 16 MB
    _Float16* att   = (_Float16*)(Gbase + (size_t)ROWS * ENC_E * 4);  // 16 MB
    _Float16* hshr  = (_Float16*)(Gbase + (size_t)ROWS * ENC_E * 4
                                        + (size_t)ROWS * (LSTM + ENC_E) * 2); // 8 MB

    const int T = 256;
    auto blocks = [](size_t n, int t) { return (int)((n + t - 1) / t); };

    // ---- weight conversion ----
    cvt_f16<<<blocks((size_t)NL * G4 * LSTM, T), T, 0, stream>>>(wih, wih16, NL * G4 * LSTM);
    cvt_f16<<<blocks((size_t)NL * G4 * LSTM, T), T, 0, stream>>>(whh, whh16, NL * G4 * LSTM);
    cvt_f16<<<blocks((size_t)ENC_E * LSTM, T), T, 0, stream>>>(attW, atw16, ENC_E * LSTM);
    cvt_f16<<<blocks((size_t)SHRINK * (LSTM + ENC_E), T), T, 0, stream>>>(shrW, shw16, SHRINK * (LSTM + ENC_E));
    cvt_outw<<<blocks((size_t)VOCABP * SHRINK, T), T, 0, stream>>>(outW, otw16);
    combine_bias<<<blocks((size_t)NL * G4, T), T, 0, stream>>>(bih, bhh, biasc, NL * G4);
    build_x0<<<blocks((size_t)ROWS * 512, T), T, 0, stream>>>(idx, embt, enci, Xa);

    // ---- 3 LSTM layers ----
    for (int l = 0; l < NL; ++l) {
        _Float16* Xin  = (l & 1) ? Xb : Xa;
        _Float16* Xout = (l & 1) ? Xa : Xb;
        init_state<<<blocks(B * LSTM, T), T, 0, stream>>>(h0 + (size_t)l * B * LSTM,
                                                          c0 + (size_t)l * B * LSTM, Hp0, Cst);
        // G = X * Wih^T + (bih+bhh)   : 8192 x 2048, K=512
        {
            int tiles = (ROWS / 32) * (G4 / 32);
            gemm_wmma<<<blocks(tiles, 8), 256, 0, stream>>>(
                Xin, wih16 + (size_t)l * G4 * LSTM, biasc + (size_t)l * G4,
                Gbuf, nullptr, ROWS, G4, LSTM, G4, G4);
        }
        for (int t = 0; t < S; ++t) {
            _Float16* Hcur = (t & 1) ? Hp1 : Hp0;
            _Float16* Hnxt = (t & 1) ? Hp0 : Hp1;
            lstm_step<<<16, 256, 0, stream>>>(
                Hcur, whh16 + (size_t)l * G4 * LSTM, Gbuf + (size_t)t * B * G4,
                Cst, Hnxt, Xout + (size_t)t * B * LSTM,
                out + LOGITS_N + (size_t)l * B * LSTM,
                out + LOGITS_N + (size_t)NL * B * LSTM + (size_t)l * B * LSTM,
                t == S - 1 ? 1 : 0);
        }
    }
    _Float16* dec = Xb;   // layer 2 output (l=2 writes Xb)

    // ---- attention ----
    {   // q = dec * attn_W^T + attn_b  (fp32 out)
        int tiles = (ROWS / 32) * (ENC_E / 32);
        gemm_wmma<<<blocks(tiles, 8), 256, 0, stream>>>(
            dec, atw16, attb, qbuf, nullptr, ROWS, ENC_E, LSTM, ENC_E, ENC_E);
    }
    attn_kernel<<<ROWS / 8, 256, 0, stream>>>(enco, qbuf, dec, att);

    // ---- shrink:  h = att * shrink_W^T + shrink_b  (f16 out, K=1024) ----
    {
        int tiles = (ROWS / 32) * (SHRINK / 32);
        gemm_wmma<<<blocks(tiles, 8), 256, 0, stream>>>(
            att, shw16, shrb, nullptr, hshr, ROWS, SHRINK, LSTM + ENC_E, SHRINK, SHRINK);
    }
    // ---- logits: out = h * out_W^T + out_b  (fp32 to d_out, ldd=10000) ----
    {
        int tiles = (ROWS / 32) * (VOCABP / 32);
        gemm_wmma<<<blocks(tiles, 8), 256, 0, stream>>>(
            hshr, otw16, outb, out, nullptr, ROWS, VOCABP, SHRINK, VOCAB, VOCAB);
    }
}
